// DenseCRF_35579509080386
// MI455X (gfx1250) — matrix operand order repaired and checked by reading, not compiled, not run on hardware
//
#include <hip/hip_runtime.h>
#include <hip/hip_bf16.h>

typedef __bf16 v16bf __attribute__((ext_vector_type(16)));
typedef __bf16 v8bf  __attribute__((ext_vector_type(8)));
typedef float  v8f   __attribute__((ext_vector_type(8)));

namespace crf {
constexpr int B  = 2;
constexpr int C  = 21;
constexpr int H  = 80;
constexpr int W  = 80;
constexpr int N  = H * W;        // 6400
constexpr int CP = 32;           // channels padded to two 16-row WMMA tiles
constexpr int KT = 32;           // K per WMMA
constexpr int NT = N / 16;       // 400 column tiles
constexpr int ITERS = 5;
constexpr float INV_2S2 = 1.0f / 18.0f;   // 1 / (2 * 3^2)
constexpr float INV_2B2 = 1.0f / 50.0f;   // 1 / (2 * 5^2)
}

// ---------------------------------------------------------------------------
// Kernel 1: build combined kernel matrix Kc[b][i][j] =
//   3*exp(-d2_spatial/18) + 5*exp(-d2_color/50), stored bf16 row-major.
// Symmetric, so row-major storage doubles as the transposed (B-side) view.
// One thread computes 8 contiguous columns (one v8bf store). 80 % 8 == 0, so
// an 8-block never straddles an image row -> yj is uniform per thread.
// ---------------------------------------------------------------------------
__global__ void build_kc(const float* __restrict__ img, __bf16* __restrict__ kc)
{
    using namespace crf;
    long tid = (long)blockIdx.x * blockDim.x + threadIdx.x;
    const long total = (long)B * N * (N / 8);
    if (tid >= total) return;

    int jblk = (int)(tid % (N / 8));
    int i    = (int)((tid / (N / 8)) % N);
    int b    = (int)(tid / ((long)N * (N / 8)));
    int j0   = jblk * 8;

    float yi  = (float)(i / W);
    float xi  = (float)(i % W);
    float yj  = (float)(j0 / W);
    float xj0 = (float)(j0 % W);
    float dy  = yi - yj;
    float dy2 = dy * dy;

    const float* im = img + (long)b * 3 * N;
    float r0 = im[i], g0 = im[N + i], b0 = im[2 * N + i];

    v8bf outv;
#pragma unroll
    for (int t = 0; t < 8; ++t) {
        int j = j0 + t;
        float dx  = xi - (xj0 + (float)t);
        float d2s = dy2 + dx * dx;
        float dr = r0 - im[j];
        float dg = g0 - im[N + j];
        float db = b0 - im[2 * N + j];
        float d2b = dr * dr + dg * dg + db * db;
        float v = 3.0f * __expf(-d2s * INV_2S2) + 5.0f * __expf(-d2b * INV_2B2);
        outv[t] = (__bf16)v;
    }
    *reinterpret_cast<v8bf*>(kc + (long)b * N * N + (long)i * N + j0) = outv;
}

// ---------------------------------------------------------------------------
// Kernel 2: pack predictions into padded bf16 Q buffers [B][CP][N];
// rows C..CP-1 are zero (WMMA padding) in both ping-pong buffers.
// ---------------------------------------------------------------------------
__global__ void pack_q(const float* __restrict__ pred,
                       __bf16* __restrict__ qa, __bf16* __restrict__ qb)
{
    using namespace crf;
    int tid = blockIdx.x * blockDim.x + threadIdx.x;
    if (tid >= B * CP * N) return;
    int n = tid % N;
    int c = (tid / N) % CP;
    int b = tid / (CP * N);
    float v = (c < C) ? pred[((long)b * C + c) * N + n] : 0.0f;
    __bf16 h = (__bf16)v;
    qa[tid] = h;
    qb[tid] = h;
}

// ---------------------------------------------------------------------------
// Kernel 3: one mean-field iteration.
//   msg[b][c][m] = sum_n Qin[b][c][n] * Kc[b][m][n]   (Kc symmetric)
//   Qout = sigmoid(pred - msg)  (bf16; fp32 to d_out on last iteration)
// One wave32 owns one 16x16 output tile, full K loop of 200 WMMA steps.
// A and B both use contiguous k order (element s holds k = 16*half + s).
// ---------------------------------------------------------------------------
__global__ void crf_step(const __bf16* __restrict__ qin,
                         __bf16* __restrict__ qout,
                         const __bf16* __restrict__ kc,
                         const float* __restrict__ pred,
                         float* __restrict__ out,
                         int writeOut)
{
    using namespace crf;
    int wave = (int)((blockIdx.x * blockDim.x + threadIdx.x) >> 5);
    int lane = threadIdx.x & 31;
    if (wave >= B * 2 * NT) return;           // grid is exact; uniform guard

    int ntile = wave % NT;
    int mtile = (wave / NT) % 2;
    int b     = wave / (NT * 2);

    int  l15 = lane & 15;
    bool hi  = lane >= 16;

    // A (16x32 bf16): lanes 0-15 hold row M=l15, K 0..15;
    //                 lanes 16-31 hold row M=l15, K 16..31.
    int arow  = mtile * 16 + l15;
    int kAoff = hi ? 16 : 0;
    // B (32x16 bf16): lanes 0-15 hold col N=l15, K 0..15; lanes 16-31, K 16..31.
    int ncol  = ntile * 16 + l15;
    int kBoff = hi ? 16 : 0;

    const __bf16* aptr = qin + ((long)b * CP + arow) * N;
    const __bf16* bptr = kc  + ((long)b * N + ncol) * N;

    v8f acc = {};
    for (int k0 = 0; k0 < N; k0 += KT) {
        v8bf alo = *reinterpret_cast<const v8bf*>(aptr + k0 + kAoff);
        v8bf ahi = *reinterpret_cast<const v8bf*>(aptr + k0 + kAoff + 8);
        v16bf a;
#pragma unroll
        for (int e = 0; e < 8; ++e) { a[e] = alo[e]; a[e + 8] = ahi[e]; }

        v16bf bm = *reinterpret_cast<const v16bf*>(bptr + k0 + kBoff);

        acc = __builtin_amdgcn_wmma_f32_16x16x32_bf16(
            /*neg_a=*/false, a, /*neg_b=*/false, bm,
            /*c_mod=*/(short)0, acc, /*reuse_a=*/false, /*reuse_b=*/false);
    }

    // D (16x16 f32): element r -> row M = r + (hi?8:0), col N = l15.
    int m = ntile * 16 + l15;
#pragma unroll
    for (int r = 0; r < 8; ++r) {
        int c = mtile * 16 + r + (hi ? 8 : 0);
        if (c < C) {
            float p = pred[((long)b * C + c) * N + m];
            float q = 1.0f / (1.0f + __expf(acc[r] - p));   // sigmoid(p - msg)
            qout[((long)b * CP + c) * N + m] = (__bf16)q;
            if (writeOut)
                out[((long)b * C + c) * N + m] = q;
        }
    }
}

// ---------------------------------------------------------------------------
extern "C" void kernel_launch(void* const* d_in, const int* in_sizes, int n_in,
                              void* d_out, int out_size, void* d_ws, size_t ws_size,
                              hipStream_t stream)
{
    using namespace crf;
    (void)in_sizes; (void)n_in; (void)out_size; (void)ws_size;

    const float* pred = (const float*)d_in[0];   // [2,21,80,80] f32
    const float* img  = (const float*)d_in[1];   // [2,3,80,80]  f32
    float* out = (float*)d_out;                  // [2,21,80,80] f32

    // Workspace layout:
    //   Kc : B*N*N bf16  = 163,840,000 bytes (fits the 192 MB L2 -> all 5
    //        GEMM iterations stream the kernel matrix from L2, not HBM)
    //   QA : B*CP*N bf16 = 819,200 bytes
    //   QB : B*CP*N bf16
    char* ws = (char*)d_ws;
    __bf16* kc = (__bf16*)ws;
    size_t kcBytes = (size_t)B * N * N * sizeof(__bf16);
    size_t qBytes  = (size_t)B * CP * N * sizeof(__bf16);
    __bf16* qa = (__bf16*)(ws + kcBytes);
    __bf16* qb = (__bf16*)(ws + kcBytes + qBytes);

    long buildThreads = (long)B * N * (N / 8);           // 10,240,000
    build_kc<<<(unsigned)((buildThreads + 255) / 256), 256, 0, stream>>>(img, kc);

    int packThreads = B * CP * N;                         // 409,600
    pack_q<<<(packThreads + 255) / 256, 256, 0, stream>>>(pred, qa, qb);

    int gemmThreads = B * 2 * NT * 32;                    // 51,200 -> 1600 waves
    __bf16* cur = qa;
    __bf16* nxt = qb;
    for (int it = 0; it < ITERS; ++it) {
        crf_step<<<(gemmThreads + 255) / 256, 256, 0, stream>>>(
            cur, nxt, kc, pred, out, it == ITERS - 1 ? 1 : 0);
        __bf16* t = cur; cur = nxt; nxt = t;
    }
}
